// Module_9182640079161
// MI455X (gfx1250) — matrix operand order repaired and checked by reading, not compiled, not run on hardware
//
#include <hip/hip_runtime.h>
#include <hip/hip_bf16.h>
#include <stdint.h>

// ---------------- problem constants ----------------
#define N_USERS 100000
#define N_ITEMS 100000
#define DDIM    128
#define LHIST   200
#define BATCH   4096
#define LPAD    208          // 13 tiles of 16 rows
#define KVS     136          // padded row stride in halves: 272B = 17*16B (aligned, bank-shifted)
#define TAU_INV 0.25f
#define SIG_Q   0.1f
#define BETA    0.25f
#define LOG_SPSQ 2.3025850929940457f   // log(SIG_P/SIG_Q) = log(10)

typedef _Float16 v16h __attribute__((ext_vector_type(16)));
typedef _Float16 v4h  __attribute__((ext_vector_type(4)));
typedef float    v8f  __attribute__((ext_vector_type(8)));

// LDS carve (bytes)
#define SKV_BYTES   (LPAD * KVS * 2)           // 56576
#define SWKT_BYTES  (DDIM * KVS * 2)           // 34816
#define SFLOATS     (8*DDIM + 3*LPAD + 256)    // Q,Tgt,QW,Vatt,CtxU,CtxI,Usl,Isl + S,Wu,Wi + Red
#define SMEM_BYTES  (SKV_BYTES + SWKT_BYTES + SFLOATS*4 + LPAD*4)

// ---------------- fast branch-free device math ----------------
__device__ __forceinline__ float fast_tanh(float x) {
    x = fminf(fmaxf(x, -15.0f), 15.0f);
    float t = __expf(2.0f * x);                       // native v_exp_f32 path
    return (t - 1.0f) * __builtin_amdgcn_rcpf(t + 1.0f);
}

// ---------------- deterministic RNG (hash -> Box-Muller) ----------------
__device__ __forceinline__ unsigned h32(unsigned x) {
    x ^= x >> 17; x *= 0xed5ad4bbu;
    x ^= x >> 11; x *= 0xac4c1b51u;
    x ^= x >> 15; x *= 0x31848babu;
    x ^= x >> 14; return x;
}
__device__ __forceinline__ float gaussr(unsigned idx, unsigned salt) {
    unsigned a = h32(idx * 2654435761u + salt);
    unsigned c = h32(a ^ 0x9e3779b9u);
    float u1 = (float)(a >> 8) * (1.0f / 16777216.0f) + 1e-7f;
    float u2 = (float)(c >> 8) * (1.0f / 16777216.0f);
    return sqrtf(-2.0f * __logf(u1)) * __cosf(6.2831853071795864f * u2);
}

// ---------------- block reductions (all 256 threads must call) ----------------
__device__ __forceinline__ float bsum(float v, float* red, int tid) {
    red[tid] = v; __syncthreads();
    #pragma unroll
    for (int st = 128; st > 0; st >>= 1) {
        if (tid < st) red[tid] += red[tid + st];
        __syncthreads();
    }
    float r = red[0]; __syncthreads();
    return r;
}
__device__ __forceinline__ float bmax(float v, float* red, int tid) {
    red[tid] = v; __syncthreads();
    #pragma unroll
    for (int st = 128; st > 0; st >>= 1) {
        if (tid < st) red[tid] = fmaxf(red[tid], red[tid + st]);
        __syncthreads();
    }
    float r = red[0]; __syncthreads();
    return r;
}

__global__ void init_acc_kernel(float* acc) {
    if (threadIdx.x < 2) acc[threadIdx.x] = 0.0f;
}

__global__ void finalize_kernel(const float* acc, float* out) {
    if (threadIdx.x == 0)
        out[BATCH] = BETA * acc[0] / fmaxf(acc[1], 1.0f);
}

__global__ __launch_bounds__(256)
void bam_fused_kernel(const float* __restrict__ user_emb,
                      const float* __restrict__ item_emb,
                      const float* __restrict__ Wq,
                      const float* __restrict__ Wk,
                      const float* __restrict__ b_att,
                      const float* __restrict__ v_att,
                      const float* __restrict__ ln_u_g,
                      const float* __restrict__ ln_u_b,
                      const float* __restrict__ ln_i_g,
                      const float* __restrict__ ln_i_b,
                      const float* __restrict__ pred_W,
                      const float* __restrict__ pred_b,
                      const int*   __restrict__ user_hist,
                      const int*   __restrict__ user_idx,
                      const int*   __restrict__ item_idx,
                      float* __restrict__ out,
                      float* __restrict__ gacc) {
    extern __shared__ char smem[];
    _Float16* sKV  = (_Float16*)smem;                       // LPAD x KVS
    _Float16* sWkT = sKV + LPAD * KVS;                      // DDIM x KVS (transposed Wk)
    float* sQ    = (float*)(smem + SKV_BYTES + SWKT_BYTES);
    float* sTgt  = sQ    + DDIM;
    float* sQW   = sTgt  + DDIM;
    float* sVatt = sQW   + DDIM;
    float* sCtxU = sVatt + DDIM;
    float* sCtxI = sCtxU + DDIM;
    float* sUsl  = sCtxI + DDIM;
    float* sIsl  = sUsl  + DDIM;
    float* sS    = sIsl  + DDIM;    // LPAD
    float* sWu   = sS    + LPAD;    // LPAD
    float* sWi   = sWu   + LPAD;    // LPAD
    float* sRed  = sWi   + LPAD;    // 256
    int*   sHist = (int*)(sRed + 256);  // LPAD

    const int tid = threadIdx.x;
    const int b   = blockIdx.x;
    const int uidx = user_idx[b];
    const int tgt  = item_idx[b];

    // ---- async gather of the history-id row straight into LDS (ASYNCcnt path) ----
    if (tid < LHIST) {
        unsigned dsoff = (unsigned)(uintptr_t)(&sHist[tid]);
        const int* gp = user_hist + (size_t)uidx * LHIST + tid;
        asm volatile("global_load_async_to_lds_b32 %0, %1, off"
                     :: "v"(dsoff), "v"(gp) : "memory");
    }

    // ---- stage Q row, target item row, v_att (overlaps with async hist fetch) ----
    if (tid < DDIM) {
        sQ[tid]    = user_emb[(size_t)uidx * DDIM + tid];
        sTgt[tid]  = item_emb[(size_t)tgt  * DDIM + tid];
        sVatt[tid] = v_att[tid];
    }
    // ---- Wk -> LDS, transposed f16 (vectorized global reads) ----
    for (int i = tid; i < DDIM * (DDIM / 4); i += 256) {
        int k = i >> 5, n = (i & 31) * 4;
        const float4 w = *(const float4*)(Wk + k * DDIM + n);
        sWkT[(n + 0) * KVS + k] = (_Float16)w.x;
        sWkT[(n + 1) * KVS + k] = (_Float16)w.y;
        sWkT[(n + 2) * KVS + k] = (_Float16)w.z;
        sWkT[(n + 3) * KVS + k] = (_Float16)w.w;
    }
    asm volatile("s_wait_asynccnt 0x0" ::: "memory");
    __syncthreads();

    // ---- gather KV -> LDS f16, float4 loads / b64 stores (rows >= LHIST zeroed) ----
    for (int i = tid; i < LPAD * (DDIM / 4); i += 256) {
        int l = i >> 5, c = (i & 31) * 4;
        float4 v = make_float4(0.0f, 0.0f, 0.0f, 0.0f);
        if (l < LHIST) v = *(const float4*)(item_emb + (size_t)sHist[l] * DDIM + c);
        v4h h; h[0] = (_Float16)v.x; h[1] = (_Float16)v.y;
               h[2] = (_Float16)v.z; h[3] = (_Float16)v.w;
        *(v4h*)(&sKV[l * KVS + c]) = h;                 // 8B aligned: 272%8==0, 2c%8==0
    }

    // ---- qw[n] = b_att[n] + sum_d Q[d]*Wq[d][n] (split d over two half-blocks) ----
    {
        const int n = tid & 127, h = tid >> 7;
        float part = 0.0f;
        #pragma unroll 4
        for (int it = 0; it < 64; ++it) {
            int d = h * 64 + it;
            part = fmaf(sQ[d], Wq[d * DDIM + n], part);
        }
        sRed[tid] = part;
        __syncthreads();
        if (tid < DDIM) sQW[tid] = sRed[tid] + sRed[tid + 128] + b_att[tid];
    }
    __syncthreads();

    // ---- WMMA phase: s[l] = (1/TAU) * sum_n tanh(qw[n] + (K@Wk)[l,n]) * v_att[n] ----
    {
        const int wave = tid >> 5;
        const int lane = tid & 31;
        const int ln   = lane & 15;
        const int hi   = lane >> 4;
        for (int lt = wave; lt < LPAD / 16; lt += 8) {   // each L-tile owned by one wave
            float acc[8];
            #pragma unroll
            for (int r = 0; r < 8; ++r) acc[r] = 0.0f;
            const int row = lt * 16 + ln;
            for (int nt = 0; nt < 8; ++nt) {
                v8f c = {};
                #pragma unroll
                for (int kc = 0; kc < DDIM; kc += 32) {
                    // A: 16x32 f16, per ISA layout (k-blocks of 8, +16 stride)
                    const _Float16* pa = sKV + row * KVS + kc + hi * 8;
                    v16h a;
                    #pragma unroll
                    for (int j = 0; j < 8; ++j) { a[j] = pa[j]; a[j + 8] = pa[j + 16]; }
                    // B: 32x16 f16 from transposed Wk (k-blocks of 8, +16 stride, same as A)
                    const _Float16* pb = sWkT + (nt * 16 + ln) * KVS + kc + hi * 8;
                    v16h bm;
                    #pragma unroll
                    for (int j = 0; j < 8; ++j) { bm[j] = pb[j]; bm[j + 8] = pb[j + 16]; }
                    c = __builtin_amdgcn_wmma_f32_16x16x32_f16(
                            false, a, false, bm, (short)0, c, false, false);
                }
                const int n = nt * 16 + ln;
                const float qwn = sQW[n], va = sVatt[n];
                #pragma unroll
                for (int r = 0; r < 8; ++r)
                    acc[r] += fast_tanh(c[r] + qwn) * va;
            }
            #pragma unroll
            for (int r = 0; r < 8; ++r) {
                float v = acc[r];
                #pragma unroll
                for (int m = 1; m < 16; m <<= 1) v += __shfl_xor(v, m, 32);
                if (ln == 0) sS[lt * 16 + r + 8 * hi] = v * TAU_INV;
            }
        }
    }
    __syncthreads();

    // ---- masked softmax + lognormal sampling + KL ----
    const int l = tid;
    const bool valid = (l < LHIST) && (sHist[l] != tgt) && (sHist[l] != N_ITEMS);
    float sv = valid ? sS[l] : -1e30f;
    float mx = bmax(sv, sRed, tid);
    float e  = valid ? __expf(sv - mx) : 0.0f;
    float sumE = bsum(e, sRed, tid) + 1e-30f;
    float nv   = bsum(valid ? 1.0f : 0.0f, sRed, tid);
    float mu   = valid ? __logf(e / sumE + 1e-24f) : 0.0f;
    float mu_p = -__logf(fmaxf(nv, 1.0f));

    unsigned gid = (unsigned)b * LHIST + (unsigned)l;
    float wu = valid ? __expf(mu + SIG_Q * gaussr(gid, 0x13572468u)) : 0.0f;
    float wi = valid ? __expf(mu + SIG_Q * gaussr(gid, 0x8badf00du)) : 0.0f;
    if (l < LHIST) { sWu[l] = wu; sWi[l] = wi; }
    float sumWu = bsum(wu, sRed, tid);
    float sumWi = bsum(wi, sRed, tid);
    float dm = mu - mu_p;
    float klv = valid ? (LOG_SPSQ + 0.5f * (SIG_Q * SIG_Q + dm * dm) - 0.5f) : 0.0f;
    float klSum = bsum(klv, sRed, tid);
    if (tid == 0) {
        atomicAdd(&gacc[0], klSum);
        atomicAdd(&gacc[1], nv);
    }

    // ---- two sampled contexts, fused GMF pre-products ----
    {
        const float invU = 1.0f / (sumWu + 1e-12f);
        const float invI = 1.0f / (sumWi + 1e-12f);
        if (tid < DDIM) {
            float s = 0.0f;
            for (int ll = 0; ll < LHIST; ++ll)
                s = fmaf(sWu[ll], (float)sKV[ll * KVS + tid], s);
            sCtxU[tid] = s * invU * sQ[tid];          // ctx_u * user_emb  (pre-LN)
        } else {
            const int d = tid - DDIM;
            float s = 0.0f;
            for (int ll = 0; ll < LHIST; ++ll)
                s = fmaf(sWi[ll], (float)sKV[ll * KVS + d], s);
            sCtxI[d] = s * invI * sTgt[d];            // ctx_i * item_emb  (pre-LN)
        }
    }
    __syncthreads();

    // ---- dual layernorm (halves of the block handle u / i independently) ----
    {
        const int d = tid & 127;
        float x = (tid < DDIM) ? sCtxU[d] : sCtxI[d];
        sRed[tid] = x; __syncthreads();
        #pragma unroll
        for (int st = 64; st > 0; st >>= 1) {
            if ((tid & 127) < st) sRed[tid] += sRed[tid + st];
            __syncthreads();
        }
        float mean = ((tid < DDIM) ? sRed[0] : sRed[128]) * (1.0f / DDIM);
        __syncthreads();
        float dx = x - mean;
        sRed[tid] = dx * dx; __syncthreads();
        #pragma unroll
        for (int st = 64; st > 0; st >>= 1) {
            if ((tid & 127) < st) sRed[tid] += sRed[tid + st];
            __syncthreads();
        }
        float var = ((tid < DDIM) ? sRed[0] : sRed[128]) * (1.0f / DDIM);
        __syncthreads();
        float g  = (tid < DDIM) ? ln_u_g[d] : ln_i_g[d];
        float bb = (tid < DDIM) ? ln_u_b[d] : ln_i_b[d];
        float sl = dx * rsqrtf(var + 1e-5f) * g + bb;
        if (tid < DDIM) sUsl[d] = sl; else sIsl[d] = sl;
    }
    __syncthreads();

    // ---- prediction head ----
    float p = (tid < DDIM) ? sUsl[tid] * sIsl[tid] * pred_W[tid] : 0.0f;
    float logit = bsum(p, sRed, tid);
    if (tid == 0) out[b] = logit + pred_b[0];
}

extern "C" void kernel_launch(void* const* d_in, const int* in_sizes, int n_in,
                              void* d_out, int out_size, void* d_ws, size_t ws_size,
                              hipStream_t stream) {
    const float* user_emb = (const float*)d_in[0];
    const float* item_emb = (const float*)d_in[1];
    const float* Wq       = (const float*)d_in[2];
    const float* Wk       = (const float*)d_in[3];
    const float* b_att    = (const float*)d_in[4];
    const float* v_att    = (const float*)d_in[5];
    const float* ln_u_g   = (const float*)d_in[6];
    const float* ln_u_b   = (const float*)d_in[7];
    const float* ln_i_g   = (const float*)d_in[8];
    const float* ln_i_b   = (const float*)d_in[9];
    const float* pred_W   = (const float*)d_in[10];
    const float* pred_b   = (const float*)d_in[11];
    const int*   user_hist = (const int*)d_in[12];
    const int*   user_idx  = (const int*)d_in[13];
    const int*   item_idx  = (const int*)d_in[14];
    float* out  = (float*)d_out;
    float* gacc = (float*)d_ws;   // [0]=kl sum, [1]=valid count

    init_acc_kernel<<<1, 32, 0, stream>>>(gacc);
    bam_fused_kernel<<<BATCH, 256, SMEM_BYTES, stream>>>(
        user_emb, item_emb, Wq, Wk, b_att, v_att,
        ln_u_g, ln_u_b, ln_i_g, ln_i_b, pred_W, pred_b,
        user_hist, user_idx, item_idx, out, gacc);
    finalize_kernel<<<1, 32, 0, stream>>>(gacc, out);
}
